// LSTMModel_43155831390313
// MI455X (gfx1250) — matrix operand order repaired and checked by reading, not compiled, not run on hardware
//
#include <hip/hip_runtime.h>

typedef __attribute__((ext_vector_type(16))) _Float16 v16h;
typedef __attribute__((ext_vector_type(8)))  float    v8f;

#define B_    256
#define T_    2048
#define NINP_ 64
#define H_    10
#define NGATE 4
#define TILE_ROWS 16
#define NTILES (B_ / TILE_ROWS)          // 16 independent batch tiles
#define NLAYER 5
#define CHUNK_ 32                        // progress publish granularity (steps)
#define PROG_STRIDE 32                   // ints; pad counters to 128B lines
#define PROG_INTS (NTILES * NLAYER * PROG_STRIDE)

// Packed f16 weight images (K-major rows of 64 padded N columns: N = gate*16 + j)
#define WIH0_ELTS (64 * 64)              // layer-0 input proj, K=64
#define WIHR_ELTS (4 * 32 * 64)          // layers 1..4 input proj, K padded to 32
#define WHH_ELTS  (5 * 32 * 64)          // recurrent, K padded to 32
#define WPACK_ELTS (WIH0_ELTS + WIHR_ELTS + WHH_ELTS)

// One h-sequence buffer: [tile][t][M=16 rows][N=16 cols] f16 = 16 MB
#define HSEQ_ELTS ((size_t)NTILES * T_ * 256)

// ---- fast activations: single v_exp_f32 + single v_rcp_f32 (no precise div) ---------
__device__ __forceinline__ float sig_(float x) {
    return __builtin_amdgcn_rcpf(1.0f + __expf(-x));
}
__device__ __forceinline__ float tanh_(float x) {
    return 1.0f - 2.0f * __builtin_amdgcn_rcpf(1.0f + __expf(2.0f * x));
}

// ---- cross-block progress sync (pipelined mode) -------------------------------------
__device__ __forceinline__ void publish_(int* prog, int v) {
    if (prog) __hip_atomic_store(prog, v, __ATOMIC_RELEASE, __HIP_MEMORY_SCOPE_AGENT);
}
__device__ __forceinline__ int wait_avail_(int* prog, int need, int avail) {
    if (!prog) return T_;
    while (avail < need) {
        avail = __hip_atomic_load(prog, __ATOMIC_ACQUIRE, __HIP_MEMORY_SCOPE_AGENT);
        if (avail < need) __builtin_amdgcn_s_sleep(2);
    }
    return avail;
}

// ---- fragment builders (wave32 CDNA5 WMMA layouts) ----------------------------------
// A (16x32 f16): lane l holds row M=l%16; element e -> K = (e<8 ? e : 16+(e-8)) + 8*(l/16)

// A-fragment from a 16-row x 16-col f16 tile (LDS hbuf or global hseq row). K>=16 zero.
__device__ __forceinline__ v16h afrag_row16(const _Float16* src, int m, int lg) {
    v16h a;
    const _Float16* p = src + m * 16 + lg * 8;   // lanes<16: cols 0..7, lanes>=16: cols 8..15
#pragma unroll
    for (int e = 0; e < 8; ++e) a[e] = p[e];
#pragma unroll
    for (int e = 8; e < 16; ++e) a[e] = (_Float16)0.0f;
    return a;
}

// A-fragment (chunk kc of K=64) from an f32 x-row, converted to f16 on the fly.
__device__ __forceinline__ v16h afrag_xrow(const float* xr, int kc, int lg) {
    v16h a;
    const float* p0 = xr + kc * 32 + lg * 8;        // e<8 : K = kc*32 + 8*lg + e
    const float* p1 = xr + kc * 32 + 16 + lg * 8;   // e>=8: K = kc*32 + 16 + 8*lg + (e-8)
#pragma unroll
    for (int e = 0; e < 8; ++e) a[e] = (_Float16)p0[e];
#pragma unroll
    for (int e = 0; e < 8; ++e) a[8 + e] = (_Float16)p1[e];
    return a;
}

// B (32x16 f16): lane l holds col N=l%16; element e -> K = (e<8 ? e : 16+(e-8)) + 8*(l/16); K-base kcbase.
__device__ __forceinline__ v16h bfrag(const _Float16* W, int kcbase, int ntile, int lane) {
    const int n  = (lane & 15) + ntile * 16;
    const int kb = kcbase + ((lane >> 4) << 3);
    v16h b;
#pragma unroll
    for (int e = 0; e < 16; ++e) b[e] = W[(size_t)(kb + e + (e >= 8 ? 8 : 0)) * 64 + n];
    return b;
}

// ---- weight repack + progress-counter init ------------------------------------------
__global__ void pack_weights_kernel(const float* __restrict__ wih0,
                                    const float* __restrict__ wihr,
                                    const float* __restrict__ whh,
                                    _Float16* __restrict__ wpack,
                                    int* __restrict__ prog) {
    int i = blockIdx.x * blockDim.x + threadIdx.x;
    if (prog && i < PROG_INTS) prog[i] = 0;
    if (i >= WPACK_ELTS) return;
    _Float16 val = (_Float16)0.0f;
    if (i < WIH0_ELTS) {
        int k = i >> 6, n = i & 63, g = n >> 4, j = n & 15;
        if (j < H_) val = (_Float16)wih0[(g * H_ + j) * NINP_ + k];
    } else if (i < WIH0_ELTS + WIHR_ELTS) {
        int r = i - WIH0_ELTS;
        int l = r / (32 * 64), rr = r % (32 * 64);
        int k = rr >> 6, n = rr & 63, g = n >> 4, j = n & 15;
        if (j < H_ && k < H_) val = (_Float16)wihr[((size_t)l * 40 + g * H_ + j) * H_ + k];
    } else {
        int r = i - WIH0_ELTS - WIHR_ELTS;
        int l = r / (32 * 64), rr = r % (32 * 64);
        int k = rr >> 6, n = rr & 63, g = n >> 4, j = n & 15;
        if (j < H_ && k < H_) val = (_Float16)whh[((size_t)l * 40 + g * H_ + j) * H_ + k];
    }
    wpack[i] = val;
}

// ---- LSTM layer bodies (one wave per 16-row batch tile) ------------------------------
// Shared per-step tail: activations + c/h update + h stores (C-fragment aligned gates).
__device__ __forceinline__ void cell_update_(v8f gfr[NGATE], v8f& cst, _Float16* hbuf,
                                             _Float16* hrow, int m, int lg) {
#pragma unroll
    for (int r = 0; r < 8; ++r) {
        float iv = sig_(gfr[0][r]);
        float fv = sig_(gfr[1][r]);
        float gv = tanh_(gfr[2][r]);
        float ov = sig_(gfr[3][r]);
        float c  = fv * cst[r] + iv * gv;
        cst[r]   = c;
        _Float16 hv = (_Float16)(ov * tanh_(c));
        const int mr = r + (lg << 3);
        hbuf[mr * 16 + m] = hv;
        hrow[mr * 16 + m] = hv;
    }
}

__device__ void lstm_layer0(const float* __restrict__ x, int tile,
                            _Float16* hout, const _Float16* Wih0, const _Float16* Whh0,
                            _Float16* hbuf, int lane, int* prog_out) {
    const int lg = lane >> 4, m = lane & 15;
    v16h bih[2][NGATE], bhh[NGATE];
#pragma unroll
    for (int g = 0; g < NGATE; ++g) {
        bih[0][g] = bfrag(Wih0, 0,  g, lane);
        bih[1][g] = bfrag(Wih0, 32, g, lane);
        bhh[g]    = bfrag(Whh0, 0,  g, lane);
    }
    v8f cst;
#pragma unroll
    for (int r = 0; r < 8; ++r) cst[r] = 0.0f;
#pragma unroll
    for (int e = 0; e < 8; ++e) hbuf[lane * 8 + e] = (_Float16)0.0f;
    __syncthreads();

    const float* xr0 = x + (size_t)(tile * 16 + m) * T_ * NINP_;
    // software pipeline: x fragments for step t+1 are loaded during step t
    v16h ax0n = afrag_xrow(xr0, 0, lg);
    v16h ax1n = afrag_xrow(xr0, 1, lg);
    for (int t = 0; t < T_; ++t) {
        v16h ax0 = ax0n, ax1 = ax1n;
        if (t + 1 < T_) {
            const float* xn = xr0 + (size_t)(t + 1) * NINP_;
            __builtin_prefetch(xn + 8 * NINP_, 0, 1);    // global_prefetch_b8, ~8 steps ahead
            ax0n = afrag_xrow(xn, 0, lg);
            ax1n = afrag_xrow(xn, 1, lg);
        }
        v16h ah = afrag_row16(hbuf, m, lg);
        v8f gfr[NGATE];
#pragma unroll
        for (int g = 0; g < NGATE; ++g) {
            v8f z;
#pragma unroll
            for (int r = 0; r < 8; ++r) z[r] = 0.0f;
            v8f acc = __builtin_amdgcn_wmma_f32_16x16x32_f16(false, ax0, false, bih[0][g],
                                                             (short)0, z, false, false);
            acc     = __builtin_amdgcn_wmma_f32_16x16x32_f16(false, ax1, false, bih[1][g],
                                                             (short)0, acc, false, false);
            gfr[g]  = __builtin_amdgcn_wmma_f32_16x16x32_f16(false, ah,  false, bhh[g],
                                                             (short)0, acc, false, false);
        }
        __syncthreads();
        cell_update_(gfr, cst, hbuf, hout + (size_t)t * 256, m, lg);
        __syncthreads();
        if (prog_out && ((((t + 1) & (CHUNK_ - 1)) == 0) || t == T_ - 1))
            publish_(prog_out, t + 1);
    }
}

__device__ void lstm_layer_rest(const _Float16* Wih, const _Float16* Whh,
                                const _Float16* hin, _Float16* hout,
                                _Float16* hbuf, int lane,
                                int* prog_in, int* prog_out) {
    const int lg = lane >> 4, m = lane & 15;
    v16h bih[NGATE], bhh[NGATE];
#pragma unroll
    for (int g = 0; g < NGATE; ++g) {
        bih[g] = bfrag(Wih, 0, g, lane);
        bhh[g] = bfrag(Whh, 0, g, lane);
    }
    v8f cst;
#pragma unroll
    for (int r = 0; r < 8; ++r) cst[r] = 0.0f;
#pragma unroll
    for (int e = 0; e < 8; ++e) hbuf[lane * 8 + e] = (_Float16)0.0f;
    __syncthreads();

    int avail = wait_avail_(prog_in, 1, 0);
    // software pipeline: row t+1 is loaded during step t (also correct for hin==hout,
    // since row t+1 is read before row t is overwritten)
    v16h ain_next = afrag_row16(hin, m, lg);
    for (int t = 0; t < T_; ++t) {
        v16h ain = ain_next;
        if (t + 1 < T_) {
            avail = wait_avail_(prog_in, t + 2, avail);
            ain_next = afrag_row16(hin + (size_t)(t + 1) * 256, m, lg);
        }
        v16h ah = afrag_row16(hbuf, m, lg);
        v8f gfr[NGATE];
#pragma unroll
        for (int g = 0; g < NGATE; ++g) {
            v8f z;
#pragma unroll
            for (int r = 0; r < 8; ++r) z[r] = 0.0f;
            v8f acc = __builtin_amdgcn_wmma_f32_16x16x32_f16(false, ain, false, bih[g],
                                                             (short)0, z, false, false);
            gfr[g]  = __builtin_amdgcn_wmma_f32_16x16x32_f16(false, ah,  false, bhh[g],
                                                             (short)0, acc, false, false);
        }
        __syncthreads();
        cell_update_(gfr, cst, hbuf, hout + (size_t)t * 256, m, lg);
        __syncthreads();
        if (prog_out && ((((t + 1) & (CHUNK_ - 1)) == 0) || t == T_ - 1))
            publish_(prog_out, t + 1);
    }
}

// ---- pipelined persistent kernel: grid = 16 tiles x 5 layers -------------------------
__global__ __launch_bounds__(32) void lstm_pipe_kernel(const float* __restrict__ x,
                                                       _Float16* __restrict__ bufs,
                                                       const _Float16* __restrict__ wpack,
                                                       int* __restrict__ prog) {
    __shared__ _Float16 hbuf[256];
    const int lane  = threadIdx.x;
    const int tile  = blockIdx.x & (NTILES - 1);
    const int layer = blockIdx.x >> 4;
    const _Float16* Wih0 = wpack;
    const _Float16* WihR = wpack + WIH0_ELTS;
    const _Float16* Whh  = wpack + WIH0_ELTS + WIHR_ELTS;
    const size_t toff = (size_t)tile * T_ * 256;

    int* pout = prog + (tile * NLAYER + layer) * PROG_STRIDE;
    if (layer == 0) {
        lstm_layer0(x, tile, bufs + toff, Wih0, Whh, hbuf, lane, pout);
    } else {
        int* pin = prog + (tile * NLAYER + layer - 1) * PROG_STRIDE;
        lstm_layer_rest(WihR + (size_t)(layer - 1) * 32 * 64,
                        Whh  + (size_t)layer * 32 * 64,
                        bufs + (size_t)(layer - 1) * HSEQ_ELTS + toff,
                        bufs + (size_t)layer * HSEQ_ELTS + toff,
                        hbuf, lane, pin, (layer < NLAYER - 1) ? pout : nullptr);
    }
}

// ---- sequential fallback: 16 blocks, single in-place buffer --------------------------
__global__ __launch_bounds__(32) void lstm_seq_kernel(const float* __restrict__ x,
                                                      _Float16* __restrict__ buf,
                                                      const _Float16* __restrict__ wpack) {
    __shared__ _Float16 hbuf[256];
    const int lane = threadIdx.x;
    const int tile = blockIdx.x;
    const _Float16* Wih0 = wpack;
    const _Float16* WihR = wpack + WIH0_ELTS;
    const _Float16* Whh  = wpack + WIH0_ELTS + WIHR_ELTS;
    _Float16* bt = buf + (size_t)tile * T_ * 256;

    lstm_layer0(x, tile, bt, Wih0, Whh, hbuf, lane, nullptr);
    for (int l = 1; l < NLAYER; ++l)
        lstm_layer_rest(WihR + (size_t)(l - 1) * 32 * 64,
                        Whh  + (size_t)l * 32 * 64,
                        bt, bt, hbuf, lane, nullptr, nullptr);
}

// ---- MLP head: tanh(w2 @ (w1 @ h)), time-parallel, VALU ------------------------------
__global__ __launch_bounds__(256) void mlp_kernel(const _Float16* __restrict__ hseq,
                                                  const float* __restrict__ w1,
                                                  const float* __restrict__ w2,
                                                  float* __restrict__ out) {
    __shared__ float w1s[50 * H_];
    __shared__ float w2s[50];
    const int tid = threadIdx.x;
    for (int i = tid; i < 50 * H_; i += 256) w1s[i] = w1[i];
    if (tid < 50) w2s[tid] = w2[tid];
    __syncthreads();

    const int idx  = blockIdx.x * 256 + tid;       // b*T + t
    const int b    = idx >> 11;                    // T_ = 2048
    const int t    = idx & (T_ - 1);
    const int tile = b >> 4, r = b & 15;
    const _Float16* hp = hseq + (((size_t)tile * T_ + t) * 256 + r * 16);
    float h[H_];
#pragma unroll
    for (int j = 0; j < H_; ++j) h[j] = (float)hp[j];
    float y = 0.0f;
#pragma unroll 5
    for (int o = 0; o < 50; ++o) {
        float a = 0.0f;
#pragma unroll
        for (int j = 0; j < H_; ++j) a = fmaf(w1s[o * H_ + j], h[j], a);
        y = fmaf(w2s[o], a, y);
    }
    out[idx] = tanh_(y);
}

extern "C" void kernel_launch(void* const* d_in, const int* in_sizes, int n_in,
                              void* d_out, int out_size, void* d_ws, size_t ws_size,
                              hipStream_t stream) {
    (void)in_sizes; (void)n_in; (void)out_size;
    const float* x      = (const float*)d_in[0];
    const float* w_ih0  = (const float*)d_in[1];
    const float* w_ihr  = (const float*)d_in[2];
    const float* w_hh   = (const float*)d_in[3];
    const float* w1     = (const float*)d_in[4];
    const float* w2     = (const float*)d_in[5];
    float* out          = (float*)d_out;

    const size_t bufBytes  = HSEQ_ELTS * sizeof(_Float16);          // 16 MB
    const size_t needPipe  = NLAYER * bufBytes + WPACK_ELTS * sizeof(_Float16)
                           + PROG_INTS * sizeof(int);
    const int packGrid = (WPACK_ELTS + 255) / 256;                  // also covers PROG_INTS

    if (ws_size >= needPipe) {
        // layer-pipelined path: 5 h buffers + progress counters
        _Float16* bufs  = (_Float16*)d_ws;
        _Float16* wpack = (_Float16*)((char*)d_ws + NLAYER * bufBytes);
        int*      prog  = (int*)((char*)d_ws + NLAYER * bufBytes
                                 + WPACK_ELTS * sizeof(_Float16));
        pack_weights_kernel<<<packGrid, 256, 0, stream>>>(w_ih0, w_ihr, w_hh, wpack, prog);
        lstm_pipe_kernel<<<NTILES * NLAYER, 32, 0, stream>>>(x, bufs, wpack, prog);
        mlp_kernel<<<(B_ * T_) / 256, 256, 0, stream>>>(
            bufs + (size_t)(NLAYER - 1) * HSEQ_ELTS, w1, w2, out);
    } else {
        // sequential fallback: single in-place buffer
        _Float16* buf   = (_Float16*)d_ws;
        _Float16* wpack = (_Float16*)((char*)d_ws + bufBytes);
        pack_weights_kernel<<<packGrid, 256, 0, stream>>>(w_ih0, w_ihr, w_hh, wpack, nullptr);
        lstm_seq_kernel<<<NTILES, 32, 0, stream>>>(x, buf, wpack);
        mlp_kernel<<<(B_ * T_) / 256, 256, 0, stream>>>(buf, w1, w2, out);
    }
}
